// MyModel7_2980707304232
// MI455X (gfx1250) — matrix operand order repaired and checked by reading, not compiled, not run on hardware
//
#include <hip/hip_runtime.h>
#include <math.h>

#define B_    16
#define S_    400
#define D_    768
#define H_    512
#define OUT_  256
#define SKP_  192
#define E_    8
#define FOURH 2048
#define NE_   (E_*FOURH)   /* 16384 */

typedef __attribute__((ext_vector_type(16))) __bf16 v16bf;
typedef __attribute__((ext_vector_type(8)))  float  v8f;

// ---- WMMA fragment loaders (CDNA5 ISA 7.12.2 layouts) ----
// A (16x32 bf16): lane holds row m=l%16; K pairs kb+8*half+{0..7} and kb+16+8*half+{0..7}
__device__ __forceinline__ v16bf ldA16(const __bf16* p) {
  union { uint4 u[2]; v16bf v; } f;
  f.u[0] = *(const uint4*)(p);
  f.u[1] = *(const uint4*)(p + 16);   // +16 elements = +32 bytes
  return f.v;
}
// B (32x16 bf16, K-major source): loads 16 consecutive elements from p (unused)
__device__ __forceinline__ v16bf ldB16(const __bf16* p) {
  union { uint4 u[2]; v16bf v; } f;
  f.u[0] = *(const uint4*)(p);
  f.u[1] = *(const uint4*)(p + 8);    // +8 elements = +16 bytes
  return f.v;
}
__device__ __forceinline__ v8f wmma_bf16(v16bf a, v16bf b, v8f c) {
  return __builtin_amdgcn_wmma_f32_16x16x32_bf16(false, a, false, b, (short)0, c, false, false);
}
__device__ __forceinline__ float sigmoidf_(float x) { return 1.0f / (1.0f + __expf(-x)); }

// ---------------- conversion kernels ----------------
__global__ void cvt_plain(const float* __restrict__ src, __bf16* __restrict__ dst, int n) {
  int i = blockIdx.x * 256 + threadIdx.x;
  if (i < n) dst[i] = (__bf16)src[i];
}
// src[batch][N][K] -> dst[batch][N][K]  (bf16, K-contiguous for WMMA B operand)
__global__ void cvt_transpose(const float* __restrict__ src, __bf16* __restrict__ dst,
                              int batch, int N, int K) {
  long long idx = (long long)blockIdx.x * 256 + threadIdx.x;
  long long total = (long long)batch * N * K;
  if (idx >= total) return;
  long long nk = (long long)N * K;
  int e = (int)(idx / nk);
  long long r = idx % nk;
  int k = (int)(r / N), n = (int)(r % N);
  dst[(size_t)e * nk + (size_t)n * K + k] = (__bf16)src[(size_t)e * nk + (size_t)n * K + k];
}

// ---------------- gating + aux loss ----------------
__global__ void gate_kernel(const float* __restrict__ spk, const float* __restrict__ Wg,
                            const float* __restrict__ bg, float* __restrict__ gates,
                            float* __restrict__ loss_out) {
  __shared__ float logits[B_][E_];
  __shared__ float gsh[B_][E_];
  int tid = threadIdx.x;                 // 128 threads: (b,e) pairs
  int b = tid >> 3, e = tid & 7;
  float acc = bg[e];
  for (int k = 0; k < SKP_; ++k) acc += spk[b * SKP_ + k] * Wg[e * SKP_ + k];
  logits[b][e] = acc;
  __syncthreads();
  if (tid < B_) {
    int i0 = 0, i1 = -1;
    float v0 = -1e30f, v1 = -1e30f;
    for (int j = 0; j < E_; ++j) {
      float v = logits[tid][j];
      if (v > v0) { v1 = v0; i1 = i0; v0 = v; i0 = j; }
      else if (v > v1) { v1 = v; i1 = j; }
    }
    float e0 = 1.0f, e1 = __expf(v1 - v0);
    float s = e0 + e1;
    for (int j = 0; j < E_; ++j) gsh[tid][j] = 0.0f;
    gsh[tid][i0] = e0 / s;
    gsh[tid][i1] = e1 / s;
    for (int j = 0; j < E_; ++j) gates[tid * E_ + j] = gsh[tid][j];
  }
  __syncthreads();
  if (tid == 0) {
    float imp[E_], ld[E_];
    for (int j = 0; j < E_; ++j) {
      float si = 0.f, sl = 0.f;
      for (int bb = 0; bb < B_; ++bb) { si += gsh[bb][j]; sl += (gsh[bb][j] > 0.f) ? 1.f : 0.f; }
      imp[j] = si; ld[j] = sl;
    }
    float loss = 0.f;
    for (int pass = 0; pass < 2; ++pass) {
      const float* v = pass ? ld : imp;
      float mean = 0.f;
      for (int j = 0; j < E_; ++j) mean += v[j];
      mean /= E_;
      float var = 0.f;
      for (int j = 0; j < E_; ++j) { float d = v[j] - mean; var += d * d; }
      var /= (E_ - 1);
      loss += var / (mean * mean + 1e-10f);
    }
    loss_out[0] = loss * 0.01f;
  }
}

// ---------------- input projection: xp[s][b][e*4H+n] = x@W_ih^T + (b_ih+b_hh) ----------------
// grid (64, 400), block 128 (4 waves). Each wave: 16(M) x 64(N) tile, K=768.
__global__ void xproj_gemm(const __bf16* __restrict__ xb, const __bf16* __restrict__ wT,
                           const float* __restrict__ b_ih, const float* __restrict__ b_hh,
                           __bf16* __restrict__ xp) {
  const int lane = threadIdx.x & 31, wave = threadIdx.x >> 5;
  const int half = lane >> 4, lrow = lane & 15;
  const int mtile = blockIdx.y;
  const int n0 = blockIdx.x * 256 + wave * 64;
  const int row = mtile * 16 + lrow;
  const __bf16* aptr = xb + (size_t)row * D_ + 8 * half;
  v8f acc0 = {}, acc1 = {}, acc2 = {}, acc3 = {};
  #pragma unroll 4
  for (int kb = 0; kb < D_; kb += 32) {
    v16bf a = ldA16(aptr + kb);
    const __bf16* bptr = wT + (size_t)(n0 + lrow) * D_ + 8 * half + kb;
    v16bf b0 = ldA16(bptr);
    v16bf b1 = ldA16(bptr + 16 * D_);
    v16bf b2 = ldA16(bptr + 32 * D_);
    v16bf b3 = ldA16(bptr + 48 * D_);
    acc0 = wmma_bf16(a, b0, acc0);
    acc1 = wmma_bf16(a, b1, acc1);
    acc2 = wmma_bf16(a, b2, acc2);
    acc3 = wmma_bf16(a, b3, acc3);
  }
  const int bb = mtile / 25;                 // 25 m-tiles per batch row (400/16)
  const int sbase = (mtile % 25) * 16;
  #pragma unroll
  for (int v = 0; v < 8; ++v) {
    int m = v + 8 * half;
    size_t orow = ((size_t)(sbase + m) * B_ + bb) * NE_;
    int n;
    n = n0 + lrow;      xp[orow + n] = (__bf16)(acc0[v] + b_ih[n] + b_hh[n]);
    n = n0 + 16 + lrow; xp[orow + n] = (__bf16)(acc1[v] + b_ih[n] + b_hh[n]);
    n = n0 + 32 + lrow; xp[orow + n] = (__bf16)(acc2[v] + b_ih[n] + b_hh[n]);
    n = n0 + 48 + lrow; xp[orow + n] = (__bf16)(acc3[v] + b_ih[n] + b_hh[n]);
  }
}

// ---------------- persistent per-expert LSTM ----------------
// grid 8 (one WG per expert), 1024 threads (32 waves). Dynamic LDS:
//   hbuf: bf16 [16][520]  (16640 B)   gbuf: f32 [16][2052] (131328 B)
#define HB_PITCH 520
#define GB_PITCH 2052
#define HB_BYTES (16 * HB_PITCH * 2)
#define LSTM_SMEM (HB_BYTES + 16 * GB_PITCH * 4)

__global__ __launch_bounds__(1024, 1)
void lstm_kernel(const __bf16* __restrict__ whhT, const __bf16* __restrict__ xp,
                 __bf16* __restrict__ hs) {
  extern __shared__ char smem[];
  __bf16* hbuf = (__bf16*)smem;
  float*  gbuf = (float*)(smem + HB_BYTES);
  const int e = blockIdx.x;
  const int tid = threadIdx.x;
  const int lane = tid & 31, wave = tid >> 5;
  const int half = lane >> 4, lrow = lane & 15;
  const int n0 = wave * 64;                       // 32 waves x 64 = 2048 = 4H
  for (int i = tid; i < 16 * HB_PITCH; i += 1024) hbuf[i] = (__bf16)0.0f;
  float creg[8];
  #pragma unroll
  for (int p = 0; p < 8; ++p) creg[p] = 0.0f;
  const __bf16* wbase = whhT + (size_t)e * H_ * FOURH;
  const __bf16* arow = hbuf + lrow * HB_PITCH + 8 * half;
  __syncthreads();

  for (int t = 0; t < S_; ++t) {
    // --- gate pre-activations: h_prev @ W_hh^T (fp32 accum) ---
    v8f acc0 = {}, acc1 = {}, acc2 = {}, acc3 = {};
    #pragma unroll 4
    for (int kb = 0; kb < H_; kb += 32) {
      v16bf a = ldA16(arow + kb);                               // ds_load_b128 x2
      const __bf16* bptr = wbase + (size_t)(n0 + lrow) * H_ + 8 * half + kb;
      v16bf b0 = ldA16(bptr);
      v16bf b1 = ldA16(bptr + 16 * H_);
      v16bf b2 = ldA16(bptr + 32 * H_);
      v16bf b3 = ldA16(bptr + 48 * H_);
      acc0 = wmma_bf16(a, b0, acc0);
      acc1 = wmma_bf16(a, b1, acc1);
      acc2 = wmma_bf16(a, b2, acc2);
      acc3 = wmma_bf16(a, b3, acc3);
    }
    // add xp, stage full [16 x 2048] gates in LDS
    const __bf16* xprow = xp + (size_t)t * B_ * NE_ + e * FOURH;
    #pragma unroll
    for (int v = 0; v < 8; ++v) {
      int m = v + 8 * half;
      float* grow = gbuf + m * GB_PITCH;
      size_t xoff = (size_t)m * NE_;
      grow[n0 + lrow]      = acc0[v] + (float)xprow[xoff + n0 + lrow];
      grow[n0 + 16 + lrow] = acc1[v] + (float)xprow[xoff + n0 + 16 + lrow];
      grow[n0 + 32 + lrow] = acc2[v] + (float)xprow[xoff + n0 + 32 + lrow];
      grow[n0 + 48 + lrow] = acc3[v] + (float)xprow[xoff + n0 + 48 + lrow];
    }
    __syncthreads();
    // --- elementwise LSTM cell update (fp32 c-state in registers) ---
    #pragma unroll
    for (int p = 0; p < 8; ++p) {
      int idx = tid + p * 1024;           // 8192 = 16*512 state elements
      int bb = idx >> 9, j = idx & 511;
      const float* grow = gbuf + bb * GB_PITCH;
      float gi = grow[j], gf = grow[512 + j], gg = grow[1024 + j], go = grow[1536 + j];
      float i_ = sigmoidf_(gi);
      float f_ = sigmoidf_(gf);
      float g_ = tanhf(gg);
      float o_ = sigmoidf_(go);
      float cn = f_ * creg[p] + i_ * g_;
      creg[p] = cn;
      float hn = o_ * tanhf(cn);
      __bf16 hb = (__bf16)hn;
      hbuf[bb * HB_PITCH + j] = hb;
      hs[(((size_t)e * S_ + t) * B_ + bb) * H_ + j] = hb;
    }
    __syncthreads();
  }
}

// ---------------- gate-weighted mix: mixed[b][s][j] = sum_e g[b][e]*h_e ----------------
__global__ void combine_kernel(const __bf16* __restrict__ hs, const float* __restrict__ gates,
                               __bf16* __restrict__ mixed) {
  int idx = blockIdx.x * 256 + threadIdx.x;
  if (idx >= B_ * S_ * H_) return;
  int j = idx & 511;
  int s = (idx >> 9) % S_;
  int b = idx / (S_ * H_);
  float acc = 0.0f;
  #pragma unroll
  for (int e = 0; e < E_; ++e)
    acc += gates[b * E_ + e] * (float)hs[(((size_t)e * S_ + s) * B_ + b) * H_ + j];
  mixed[idx] = (__bf16)acc;
}

// ---------------- output FC: y = mixed @ Wf^T + bf  (M=6400, N=256, K=512) ----------------
__global__ void out_gemm(const __bf16* __restrict__ mixed, const __bf16* __restrict__ wfT,
                         const float* __restrict__ bf, float* __restrict__ y) {
  const int lane = threadIdx.x & 31, wave = threadIdx.x >> 5;
  const int half = lane >> 4, lrow = lane & 15;
  const int mtile = blockIdx.x;
  const int n0 = wave * 64;                       // 4 waves x 64 = 256
  const int row = mtile * 16 + lrow;
  const __bf16* aptr = mixed + (size_t)row * H_ + 8 * half;
  v8f acc0 = {}, acc1 = {}, acc2 = {}, acc3 = {};
  #pragma unroll 4
  for (int kb = 0; kb < H_; kb += 32) {
    v16bf a = ldA16(aptr + kb);
    const __bf16* bptr = wfT + (size_t)(n0 + lrow) * H_ + 8 * half + kb;
    v16bf b0 = ldA16(bptr);
    v16bf b1 = ldA16(bptr + 16 * H_);
    v16bf b2 = ldA16(bptr + 32 * H_);
    v16bf b3 = ldA16(bptr + 48 * H_);
    acc0 = wmma_bf16(a, b0, acc0);
    acc1 = wmma_bf16(a, b1, acc1);
    acc2 = wmma_bf16(a, b2, acc2);
    acc3 = wmma_bf16(a, b3, acc3);
  }
  #pragma unroll
  for (int v = 0; v < 8; ++v) {
    int m = v + 8 * half;
    size_t orow = (size_t)(mtile * 16 + m) * OUT_;
    int n;
    n = n0 + lrow;      y[orow + n] = acc0[v] + bf[n];
    n = n0 + 16 + lrow; y[orow + n] = acc1[v] + bf[n];
    n = n0 + 32 + lrow; y[orow + n] = acc2[v] + bf[n];
    n = n0 + 48 + lrow; y[orow + n] = acc3[v] + bf[n];
  }
}

extern "C" void kernel_launch(void* const* d_in, const int* in_sizes, int n_in,
                              void* d_out, int out_size, void* d_ws, size_t ws_size,
                              hipStream_t stream) {
  (void)in_sizes; (void)n_in; (void)out_size; (void)ws_size;
  const float* x    = (const float*)d_in[0];
  const float* spk  = (const float*)d_in[1];
  const float* Wg   = (const float*)d_in[2];
  const float* bg   = (const float*)d_in[3];
  const float* W_ih = (const float*)d_in[4];
  const float* W_hh = (const float*)d_in[5];
  const float* b_ih = (const float*)d_in[6];
  const float* b_hh = (const float*)d_in[7];
  const float* Wf   = (const float*)d_in[8];
  const float* bf   = (const float*)d_in[9];
  float* y = (float*)d_out;

  char* ws = (char*)d_ws;
  size_t off = 0;
  auto alloc = [&](size_t bytes) -> void* {
    void* p = ws + off;
    off += (bytes + 255) & ~(size_t)255;
    return p;
  };
  __bf16* x_bf   = (__bf16*)alloc((size_t)B_ * S_ * D_ * 2);      // 9.8 MB
  __bf16* WihT   = (__bf16*)alloc((size_t)D_ * NE_ * 2);          // 25.2 MB
  __bf16* WhhT   = (__bf16*)alloc((size_t)E_ * H_ * FOURH * 2);   // 16.8 MB
  __bf16* WfT    = (__bf16*)alloc((size_t)H_ * OUT_ * 2);         // 0.26 MB
  __bf16* xp     = (__bf16*)alloc((size_t)S_ * B_ * NE_ * 2);     // 209.7 MB
  __bf16* hs     = (__bf16*)alloc((size_t)E_ * S_ * B_ * H_ * 2); // 52.4 MB
  __bf16* mixed  = (__bf16*)alloc((size_t)B_ * S_ * H_ * 2);      // 6.6 MB
  float*  gates  = (float*)alloc((size_t)B_ * E_ * 4);

  const int nx = B_ * S_ * D_;
  cvt_plain<<<(nx + 255) / 256, 256, 0, stream>>>(x, x_bf, nx);
  const long long nih = (long long)NE_ * D_;
  cvt_transpose<<<(unsigned)((nih + 255) / 256), 256, 0, stream>>>(W_ih, WihT, 1, NE_, D_);
  const long long nhh = (long long)E_ * FOURH * H_;
  cvt_transpose<<<(unsigned)((nhh + 255) / 256), 256, 0, stream>>>(W_hh, WhhT, E_, FOURH, H_);
  const long long nwf = (long long)OUT_ * H_;
  cvt_transpose<<<(unsigned)((nwf + 255) / 256), 256, 0, stream>>>(Wf, WfT, 1, OUT_, H_);

  gate_kernel<<<1, 128, 0, stream>>>(spk, Wg, bg, gates, y + (size_t)B_ * S_ * OUT_);

  xproj_gemm<<<dim3(NE_ / 256, (B_ * S_) / 16), 128, 0, stream>>>(x_bf, WihT, b_ih, b_hh, xp);

  (void)hipFuncSetAttribute((const void*)lstm_kernel,
                            hipFuncAttributeMaxDynamicSharedMemorySize, LSTM_SMEM);
  lstm_kernel<<<E_, 1024, LSTM_SMEM, stream>>>(WhhT, xp, hs);

  const int nmix = B_ * S_ * H_;
  combine_kernel<<<(nmix + 255) / 256, 256, 0, stream>>>(hs, gates, mixed);

  out_gemm<<<(B_ * S_) / 16, 128, 0, stream>>>(mixed, WfT, bf, y);
}
